// BiLSTM_CRF_41214506173091
// MI455X (gfx1250) — matrix operand order repaired and checked by reading, not compiled, not run on hardware
//
#include <hip/hip_runtime.h>
#include <hip/hip_bf16.h>
#include <stdint.h>

typedef __attribute__((ext_vector_type(16))) _Float16 v16h;
typedef __attribute__((ext_vector_type(8)))  float    v8f;
typedef _Float16 h16;

union Frag { v16h v; uint32_t u[8]; };

constexpr int Bsz  = 256;
constexpr int Tlen = 512;
constexpr int EMB  = 100;
constexpr int EPAD = 128;          // K pad for input proj
constexpr int HID  = 75;
constexpr int HPAD = 96;           // K pad for recurrence
constexpr int G    = 300;          // 4*HID
constexpr int GP   = 304;          // N pad (19 tiles of 16)
constexpr int NT   = 19;
constexpr int KT   = 9;            // tags
constexpr int HS_LD = 160;         // 2*HID padded
constexpr int FE_LD = 16;          // emissions cols padded
constexpr int ROWS = Bsz * Tlen;   // 131072
constexpr int MT   = ROWS / 16;    // 8192 M-tiles

__device__ inline int laneid() { return (int)__builtin_amdgcn_mbcnt_lo(~0u, 0u); }
__device__ inline float sigf(float x) { return 1.f / (1.f + __expf(-x)); }
__device__ inline float tanhf_(float x) {
  x = fminf(fmaxf(x, -15.f), 15.f);
  float e = __expf(2.f * x);
  return (e - 1.f) / (e + 1.f);
}

// ---- WMMA fragment loaders (CDNA5 16x16x32 f16 VGPR layouts) ----
// A (16x32, M x K), row-major source with leading dim ld (elements):
//   lanes 0-15: M=lane, K pairs {0..7,16..23}; lanes 16-31: M=lane-16, K {8..15,24..31}
__device__ inline v16h loadA(const h16* base, int ld, int lane, int kb) {
  Frag f;
  int koff = (lane & 16) ? 8 : 0;
  const h16* row = base + (size_t)(lane & 15) * ld + kb;
#pragma unroll
  for (int v2 = 0; v2 < 4; ++v2)
    f.u[v2] = *(const uint32_t*)(row + koff + 2 * v2);
#pragma unroll
  for (int v2 = 0; v2 < 4; ++v2)
    f.u[4 + v2] = *(const uint32_t*)(row + 16 + koff + 2 * v2);
  return f.v;
}
// B (32x16, K x N) sourced from row-major W[N][K] (B[k][n] = W[n][k]):
//   lanes 0-15: N=lane, K {0..7,16..23}; lanes 16-31: N=lane-16, K {8..15,24..31} (pairs per VGPR)
__device__ inline v16h loadB(const h16* W, int ld, int lane, int n0, int kb) {
  Frag f;
  int koff = (lane & 16) ? 8 : 0;
  const h16* row = W + (size_t)(n0 + (lane & 15)) * ld + kb + koff;
#pragma unroll
  for (int v2 = 0; v2 < 4; ++v2)
    f.u[v2] = *(const uint32_t*)(row + 2 * v2);
#pragma unroll
  for (int v2 = 0; v2 < 4; ++v2)
    f.u[4 + v2] = *(const uint32_t*)(row + 16 + 2 * v2);
  return f.v;
}

// ---- prep: pad/convert weights to f16, fuse biases ----
__global__ __launch_bounds__(256) void prep_weights(
    const float* Wih_f, const float* Whh_f, const float* bih_f, const float* bhh_f,
    const float* Wih_b, const float* Whh_b, const float* bih_b, const float* bhh_b,
    const float* Wout, const float* bout,
    h16* Wihh_f, h16* Wihh_b, h16* Whhh_f, h16* Whhh_b, h16* Wouth,
    float* bsum_f, float* bsum_b, float* boutp) {
  int id = blockIdx.x * 256 + threadIdx.x;
  if (id < GP * EPAD) {
    int n = id / EPAD, k = id % EPAD;
    bool ok = (n < G) && (k < EMB);
    Wihh_f[id] = ok ? (h16)Wih_f[n * EMB + k] : (h16)0.0f;
    Wihh_b[id] = ok ? (h16)Wih_b[n * EMB + k] : (h16)0.0f;
  }
  if (id < GP * HPAD) {
    int n = id / HPAD, k = id % HPAD;
    bool ok = (n < G) && (k < HID);
    Whhh_f[id] = ok ? (h16)Whh_f[n * HID + k] : (h16)0.0f;
    Whhh_b[id] = ok ? (h16)Whh_b[n * HID + k] : (h16)0.0f;
  }
  if (id < 16 * HS_LD) {
    int n = id / HS_LD, k = id % HS_LD;
    bool ok = (n < KT) && (k < 2 * HID);
    Wouth[id] = ok ? (h16)Wout[n * (2 * HID) + k] : (h16)0.0f;
  }
  if (id < GP) {
    bsum_f[id] = (id < G) ? (bih_f[id] + bhh_f[id]) : 0.f;
    bsum_b[id] = (id < G) ? (bih_b[id] + bhh_b[id]) : 0.f;
  }
  if (id < FE_LD) boutp[id] = (id < KT) ? bout[id] : 0.f;
}

// ---- embedding gather -> f16 padded [ROWS, EPAD] ----
__global__ __launch_bounds__(256) void embed_f16(
    const int* sent, const float* emb, h16* Xh) {
  int id = blockIdx.x * 256 + threadIdx.x;           // over ROWS*EPAD
  int i = id >> 7, k = id & (EPAD - 1);
  int tok = sent[i];
  Xh[id] = (k < EMB) ? (h16)emb[(size_t)tok * EMB + k] : (h16)0.0f;
}

// ---- input projection: XP[ROWS,GP] (f16) = Xh @ Wih^T + (bih+bhh) ----
__global__ __launch_bounds__(256) void inproj(
    const h16* Xh, const h16* Wihh_f, const h16* Wihh_b,
    const float* bsum_f, const float* bsum_b, h16* XPf, h16* XPb) {
  int dir = blockIdx.y;
  const h16* W = dir ? Wihh_b : Wihh_f;
  const float* bias = dir ? bsum_b : bsum_f;
  h16* XP = dir ? XPb : XPf;

  int gtile = blockIdx.x * 8 + (threadIdx.x >> 5);
  if (gtile >= MT * NT) return;
  int mt = gtile / NT, nt = gtile % NT;
  int lane = laneid();
  int n = lane & 15, mbase = (lane & 16) ? 8 : 0;

  v8f c = {};
  const h16* A0 = Xh + (size_t)(mt * 16) * EPAD;
#pragma unroll
  for (int kb = 0; kb < EPAD; kb += 32) {
    v16h a = loadA(A0, EPAD, lane, kb);
    v16h b = loadB(W, EPAD, lane, nt * 16, kb);
    c = __builtin_amdgcn_wmma_f32_16x16x32_f16(false, a, false, b, (short)0, c, false, false);
  }
  float bi = bias[nt * 16 + n];
#pragma unroll
  for (int r = 0; r < 8; ++r)
    XP[(size_t)(mt * 16 + mbase + r) * GP + nt * 16 + n] = (h16)(c[r] + bi);
}

// ---- LSTM time scan: 16 batch rows per WG, WMMA recurrence + gate phase ----
__global__ __launch_bounds__(256) void lstm_scan(
    const h16* XPf, const h16* XPb, const h16* Whhh_f, const h16* Whhh_b, h16* HS) {
  __shared__ h16 h_lds[16 * HPAD];      // current hidden state, f16 padded
  __shared__ float gpre[16 * GP];       // h @ Whh^T pre-activations

  int dir = blockIdx.y;
  int b0 = blockIdx.x * 16;
  const h16* XP = dir ? XPb : XPf;
  const h16* W = dir ? Whhh_b : Whhh_f;
  int colOff = dir ? HID : 0;

  int tid = threadIdx.x;
  for (int i = tid; i < 16 * HPAD; i += 256) h_lds[i] = (h16)0.0f;
  __syncthreads();

  int wave = tid >> 5, lane = laneid();
  int n = lane & 15, mbase = (lane & 16) ? 8 : 0;

  float cstate[5];
#pragma unroll
  for (int j = 0; j < 5; ++j) cstate[j] = 0.f;

  for (int s = 0; s < Tlen; ++s) {
    int t = dir ? (Tlen - 1 - s) : s;
    // gpre = h @ Whh^T  (19 N-tiles over 8 waves, K = 3x32)
    for (int nt = wave; nt < NT; nt += 8) {
      v8f c = {};
#pragma unroll
      for (int kb = 0; kb < HPAD; kb += 32) {
        v16h a = loadA(h_lds, HPAD, lane, kb);
        v16h b = loadB(W, HPAD, lane, nt * 16, kb);
        c = __builtin_amdgcn_wmma_f32_16x16x32_f16(false, a, false, b, (short)0, c, false, false);
      }
#pragma unroll
      for (int r = 0; r < 8; ++r)
        gpre[(mbase + r) * GP + nt * 16 + n] = c[r];
    }
    __syncthreads();
    // gates + state update (c in registers, h -> LDS + global HS)
#pragma unroll
    for (int j = 0; j < 5; ++j) {
      int e = tid + 256 * j;
      if (e < 16 * HID) {
        int row = e / HID, col = e % HID;
        const h16* xp = XP + ((size_t)(b0 + row) * Tlen + t) * GP;
        float gi = gpre[row * GP + col]            + (float)xp[col];
        float gf = gpre[row * GP + HID + col]      + (float)xp[HID + col];
        float gg = gpre[row * GP + 2 * HID + col]  + (float)xp[2 * HID + col];
        float go = gpre[row * GP + 3 * HID + col]  + (float)xp[3 * HID + col];
        float cc = sigf(gf) * cstate[j] + sigf(gi) * tanhf_(gg);
        cstate[j] = cc;
        float hh = sigf(go) * tanhf_(cc);
        h_lds[row * HPAD + col] = (h16)hh;
        HS[((size_t)(b0 + row) * Tlen + t) * HS_LD + colOff + col] = (h16)hh;
      }
    }
    __syncthreads();
  }
}

// ---- emissions: feats[ROWS,16] f32 = HS @ Wout^T + bout ----
__global__ __launch_bounds__(256) void emit_feats(
    const h16* HS, const h16* Wouth, const float* boutp, float* feats) {
  int mt = blockIdx.x * 8 + (threadIdx.x >> 5);
  if (mt >= MT) return;
  int lane = laneid();
  int n = lane & 15, mbase = (lane & 16) ? 8 : 0;
  v8f c = {};
  const h16* A0 = HS + (size_t)(mt * 16) * HS_LD;
#pragma unroll
  for (int kb = 0; kb < HS_LD; kb += 32) {
    v16h a = loadA(A0, HS_LD, lane, kb);
    v16h b = loadB(Wouth, HS_LD, lane, 0, kb);
    c = __builtin_amdgcn_wmma_f32_16x16x32_f16(false, a, false, b, (short)0, c, false, false);
  }
  float bi = boutp[n];
#pragma unroll
  for (int r = 0; r < 8; ++r)
    feats[(size_t)(mt * 16 + mbase + r) * FE_LD + n] = c[r] + bi;
}

// ---- CRF NLL: one wave per batch row; 9-state logsumexp scan via shuffles ----
__global__ __launch_bounds__(256) void crf_nll(
    const float* feats, const int* tags, const float* start, const float* endt,
    const float* trans, float* scores) {
  int wave = threadIdx.x >> 5, lane = laneid();
  int b = blockIdx.x * 8 + wave;
  if (b >= Bsz) return;
  const float* fb = feats + (size_t)b * Tlen * FE_LD;
  const int* tg = tags + (size_t)b * Tlen;

  // gold-path score (lanes stride over t, wave reduce)
  float part = 0.f;
  for (int t = lane; t < Tlen; t += 32) {
    int yt = tg[t];
    part += fb[t * FE_LD + yt];
    if (t > 0) part += trans[tg[t - 1] * KT + yt];
  }
#pragma unroll
  for (int off = 16; off; off >>= 1) part += __shfl_xor(part, off, 32);
  float num = part + start[tg[0]] + endt[tg[Tlen - 1]];

  // forward algorithm
  int j = lane;
  float tcol[KT];
#pragma unroll
  for (int i = 0; i < KT; ++i) tcol[i] = (j < KT) ? trans[i * KT + j] : 0.f;
  float alpha = (j < KT) ? (start[j] + fb[j]) : -1e30f;
  for (int t = 1; t < Tlen; ++t) {
    float av[KT];
#pragma unroll
    for (int i = 0; i < KT; ++i) av[i] = __shfl(alpha, i, 32) + tcol[i];
    float m = av[0];
#pragma unroll
    for (int i = 1; i < KT; ++i) m = fmaxf(m, av[i]);
    float ss = 0.f;
#pragma unroll
    for (int i = 0; i < KT; ++i) ss += __expf(av[i] - m);
    float na = m + __logf(ss) + fb[t * FE_LD + ((j < KT) ? j : 0)];
    alpha = (j < KT) ? na : -1e30f;
  }
  float x = (j < KT) ? (alpha + endt[j]) : -1e30f;
  float mx = x;
#pragma unroll
  for (int off = 16; off; off >>= 1) mx = fmaxf(mx, __shfl_xor(mx, off, 32));
  float se = __expf(x - mx);
#pragma unroll
  for (int off = 16; off; off >>= 1) se += __shfl_xor(se, off, 32);
  float denom = mx + __logf(se);
  if (lane == 0) scores[b] = denom - num;
}

__global__ __launch_bounds__(256) void reduce_mean(const float* scores, float* out) {
  __shared__ float sm[256];
  int tid = threadIdx.x;
  sm[tid] = scores[tid];
  __syncthreads();
  for (int s = 128; s; s >>= 1) {
    if (tid < s) sm[tid] += sm[tid + s];
    __syncthreads();
  }
  if (tid == 0) out[0] = sm[0] / (float)Bsz;
}

extern "C" void kernel_launch(void* const* d_in, const int* in_sizes, int n_in,
                              void* d_out, int out_size, void* d_ws, size_t ws_size,
                              hipStream_t stream) {
  const int*   sent   = (const int*)d_in[0];
  const int*   tags   = (const int*)d_in[1];
  const float* emb    = (const float*)d_in[2];
  const float* Wih_f  = (const float*)d_in[3];
  const float* Whh_f  = (const float*)d_in[4];
  const float* bih_f  = (const float*)d_in[5];
  const float* bhh_f  = (const float*)d_in[6];
  const float* Wih_b  = (const float*)d_in[7];
  const float* Whh_b  = (const float*)d_in[8];
  const float* bih_b  = (const float*)d_in[9];
  const float* bhh_b  = (const float*)d_in[10];
  const float* Wout   = (const float*)d_in[11];
  const float* bout   = (const float*)d_in[12];
  const float* startt = (const float*)d_in[13];
  const float* endt   = (const float*)d_in[14];
  const float* trans  = (const float*)d_in[15];
  float* out = (float*)d_out;

  size_t off = 0;
  auto alloc = [&](size_t bytes) {
    void* p = (char*)d_ws + off;
    off = (off + bytes + 255) & ~(size_t)255;
    return p;
  };
  h16*   Xh     = (h16*)  alloc((size_t)ROWS * EPAD * 2);
  h16*   Wihh_f = (h16*)  alloc((size_t)GP * EPAD * 2);
  h16*   Wihh_b = (h16*)  alloc((size_t)GP * EPAD * 2);
  h16*   Whhh_f = (h16*)  alloc((size_t)GP * HPAD * 2);
  h16*   Whhh_b = (h16*)  alloc((size_t)GP * HPAD * 2);
  h16*   Wouth  = (h16*)  alloc((size_t)16 * HS_LD * 2);
  float* bsum_f = (float*)alloc((size_t)GP * 4);
  float* bsum_b = (float*)alloc((size_t)GP * 4);
  float* boutp  = (float*)alloc((size_t)FE_LD * 4);
  h16*   XPf    = (h16*)  alloc((size_t)ROWS * GP * 2);
  h16*   XPb    = (h16*)  alloc((size_t)ROWS * GP * 2);
  h16*   HS     = (h16*)  alloc((size_t)ROWS * HS_LD * 2);
  float* feats  = (float*)alloc((size_t)ROWS * FE_LD * 4);
  float* scores = (float*)alloc((size_t)Bsz * 4);

  prep_weights<<<(GP * EPAD + 255) / 256, 256, 0, stream>>>(
      Wih_f, Whh_f, bih_f, bhh_f, Wih_b, Whh_b, bih_b, bhh_b, Wout, bout,
      Wihh_f, Wihh_b, Whhh_f, Whhh_b, Wouth, bsum_f, bsum_b, boutp);

  embed_f16<<<(ROWS * EPAD) / 256, 256, 0, stream>>>(sent, emb, Xh);

  inproj<<<dim3((MT * NT) / 8, 2), 256, 0, stream>>>(
      Xh, Wihh_f, Wihh_b, bsum_f, bsum_b, XPf, XPb);

  lstm_scan<<<dim3(Bsz / 16, 2), 256, 0, stream>>>(XPf, XPb, Whhh_f, Whhh_b, HS);

  emit_feats<<<MT / 8, 256, 0, stream>>>(HS, Wouth, boutp, feats);

  crf_nll<<<Bsz / 8, 256, 0, stream>>>(feats, tags, startt, endt, trans, scores);

  reduce_mean<<<1, 256, 0, stream>>>(scores, out);
}
